// VanillaVectorQuantizer_5042291605873
// MI455X (gfx1250) — matrix operand order repaired and checked by reading, not compiled, not run on hardware
//
#include <hip/hip_runtime.h>
#include <hip/hip_bf16.h>

typedef __attribute__((ext_vector_type(16))) __bf16 v16bf;
typedef __attribute__((ext_vector_type(8)))  __bf16 v8bf;
typedef __attribute__((ext_vector_type(8)))  float  v8f;

#define RVQ_NTOK 32768
#define RVQ_EDIM 256
#define RVQ_NE   8192
#define RVQ_DEPTH 4
#define RVQ_NTILE (RVQ_NE / 16)      // 512 code tiles
#define RVQ_TILEB 8192               // bytes per 16x256 bf16 code tile

union BF16x16 { v16bf v; v8bf h[2]; };

// CDNA5 async copy: global -> LDS, 16B per lane, tracked by ASYNCcnt.
__device__ __forceinline__ void rvq_async_b128(unsigned lds, const void* g) {
  asm volatile("global_load_async_to_lds_b128 %0, %1, off"
               :: "v"(lds), "v"(g) : "memory");
}
__device__ __forceinline__ void rvq_wait_async0() {
  asm volatile("s_wait_asynccnt 0" ::: "memory");
}

// Each wave copies K-chunks 2w and 2w+1 of one 8KB code tile into LDS.
// LDS unit layout: kc*1024 + piece*512 + lane*16  (conflict-minimal reads).
__device__ __forceinline__ void rvq_issue_tile(const char* ebc, size_t gsrc_off,
                                               unsigned ldsbase, int wave, int tile) {
  const char* g = ebc + (size_t)tile * RVQ_TILEB + gsrc_off + (size_t)(wave * 2) * 64;
  unsigned l = ldsbase + (unsigned)(wave * 2) * 1024;
  rvq_async_b128(l,              g);          // kc=2w,  piece 0
  rvq_async_b128(l + 512,        g + 16);     // kc=2w,  piece 1
  rvq_async_b128(l + 1024,       g + 64);     // kc=2w+1, piece 0
  rvq_async_b128(l + 1024 + 512, g + 64 + 16);// kc=2w+1, piece 1
}

// ---------------------------------------------------------------------------
// prep_emb: fp32 codebook -> bf16 copy, plus ||e||^2 per code row.
// ---------------------------------------------------------------------------
__global__ __launch_bounds__(256)
void rvq_prep_emb(const float* __restrict__ emb, __bf16* __restrict__ eb,
                  float* __restrict__ esq) {
  const int row = blockIdx.x;
  const int tid = threadIdx.x;
  float e = emb[(size_t)row * RVQ_EDIM + tid];
  eb[(size_t)row * RVQ_EDIM + tid] = (__bf16)e;
  float s = e * e;
#pragma unroll
  for (int off = 16; off > 0; off >>= 1) s += __shfl_down(s, off, 32);
  __shared__ float ps[8];
  if ((tid & 31) == 0) ps[tid >> 5] = s;
  __syncthreads();
  if (tid == 0) {
    float t = 0.0f;
#pragma unroll
    for (int w = 0; w < 8; ++w) t += ps[w];
    esq[row] = t;
  }
}

// ---------------------------------------------------------------------------
// prep_r: residual r = z (fp32) and its bf16 mirror; zero the loss cell.
// ---------------------------------------------------------------------------
__global__ __launch_bounds__(256)
void rvq_prep_r(const float* __restrict__ z, float* __restrict__ rf,
                __bf16* __restrict__ rb, float* __restrict__ out) {
  size_t i = (size_t)blockIdx.x * 256 + threadIdx.x;
  float v = z[i];
  rf[i] = v;
  rb[i] = (__bf16)v;
  if (i == 0) out[0] = 0.0f;
}

// ---------------------------------------------------------------------------
// argmin: per wave, 32 tokens (two A tiles resident in VGPRs); all 4 waves of
// a block share B tiles staged in LDS by async copies (double buffered).
// Minimize 0.5*||e||^2 - r.e with v_wmma_f32_16x16x32_bf16.
// ---------------------------------------------------------------------------
__global__ __launch_bounds__(128)
void rvq_argmin(const __bf16* __restrict__ rb, const __bf16* __restrict__ eb,
                const float* __restrict__ esq, int* __restrict__ idx) {
  const int lane = threadIdx.x & 31;
  const int wave = threadIdx.x >> 5;
  const int lrow = lane & 15;   // row within half-wave (M for A, N for B)
  const int hf   = lane >> 4;   // half-wave selector (contiguous K-split: k = 16*hf + s)
  const int tb   = (blockIdx.x * 4 + wave) * 32;

  __shared__ __align__(16) char sm[2][RVQ_TILEB];

  // Load A: rows tb..tb+31 of r_bf16, all K=256, into registers.
  v16bf a0[8], a1[8];
  {
    const __bf16* r0 = rb + (size_t)(tb + lrow) * RVQ_EDIM + hf * 16;
    const __bf16* r1 = r0 + (size_t)16 * RVQ_EDIM;
#pragma unroll
    for (int kc = 0; kc < 8; ++kc) {
      BF16x16 t0, t1;
      t0.h[0] = *(const v8bf*)(r0 + kc * 32);
      t0.h[1] = *(const v8bf*)(r0 + kc * 32 + 8);
      t1.h[0] = *(const v8bf*)(r1 + kc * 32);
      t1.h[1] = *(const v8bf*)(r1 + kc * 32 + 8);
      a0[kc] = t0.v;
      a1[kc] = t1.v;
    }
  }

  float mn[16];
  int   mi[16];
#pragma unroll
  for (int j = 0; j < 16; ++j) { mn[j] = 3.0e38f; mi[j] = 0; }

  // Async-copy bookkeeping (per lane).
  const char* ebc = (const char*)eb;
  const size_t gsrc_off = (size_t)lrow * 512 + (size_t)hf * 32;
  const unsigned lds0 = (unsigned)(uintptr_t)&sm[0][0] + (unsigned)(lane * 16);
  const unsigned lds1 = (unsigned)(uintptr_t)&sm[1][0] + (unsigned)(lane * 16);

  // Prime the double buffer: tile 0 -> buf0 (complete), tile 1 -> buf1 (in flight).
  rvq_issue_tile(ebc, gsrc_off, lds0, wave, 0);
  rvq_wait_async0();
  __syncthreads();
  rvq_issue_tile(ebc, gsrc_off, lds1, wave, 1);

  float eh = 0.5f * esq[lrow];

  for (int t = 0; t < RVQ_NTILE; ++t) {
    const int p = t & 1;
    const char* sb = &sm[p][0] + lane * 16;
    float ehn = 0.5f * esq[(((t + 1) & (RVQ_NTILE - 1)) << 4) + lrow];

    v8f c0 = {};
    v8f c1 = {};
#pragma unroll
    for (int kc = 0; kc < 8; ++kc) {
      BF16x16 bb;
      bb.h[0] = *(const v8bf*)(sb + kc * 1024);         // piece 0 (K +0..7)
      bb.h[1] = *(const v8bf*)(sb + kc * 1024 + 512);   // piece 1 (K +8..15)
      c0 = __builtin_amdgcn_wmma_f32_16x16x32_bf16(false, a0[kc], false, bb.v,
                                                   (short)0, c0, false, false);
      c1 = __builtin_amdgcn_wmma_f32_16x16x32_bf16(false, a1[kc], false, bb.v,
                                                   (short)0, c1, false, false);
    }

    const int n = t * 16 + lrow;
#pragma unroll
    for (int j = 0; j < 8; ++j) {
      float v0 = eh - c0[j];
      if (v0 < mn[j])     { mn[j] = v0;     mi[j] = n; }
      float v1 = eh - c1[j];
      if (v1 < mn[8 + j]) { mn[8 + j] = v1; mi[8 + j] = n; }
    }
    eh = ehn;

    // Pipeline: own copies for tile t+1 done; all waves done reading buf p.
    rvq_wait_async0();
    __syncthreads();
    if (t + 2 < RVQ_NTILE)
      rvq_issue_tile(ebc, gsrc_off, p ? lds1 : lds0, wave, t + 2);
  }

  // Reduce argmin across the 16 lanes that share each output row.
#pragma unroll
  for (int off = 1; off < 16; off <<= 1) {
#pragma unroll
    for (int j = 0; j < 16; ++j) {
      float ov = __shfl_xor(mn[j], off, 16);
      int   om = __shfl_xor(mi[j], off, 16);
      if (ov < mn[j] || (ov == mn[j] && om < mi[j])) { mn[j] = ov; mi[j] = om; }
    }
  }

  // C/D layout: lanes 0-15 -> M=0..7 (vgpr j), lanes 16-31 -> M=8..15.
  if (lrow == 0) {
#pragma unroll
    for (int j = 0; j < 8; ++j) {
      idx[tb + hf * 8 + j]      = mi[j];       // tile 0 (tokens tb..tb+15)
      idx[tb + 16 + hf * 8 + j] = mi[8 + j];   // tile 1 (tokens tb+16..tb+31)
    }
  }
}

// ---------------------------------------------------------------------------
// update: r -= emb[idx[t]]  (fp32 exact), refresh bf16 mirror.
// ---------------------------------------------------------------------------
__global__ __launch_bounds__(256)
void rvq_update(const float* __restrict__ emb, const int* __restrict__ idx,
                float* __restrict__ rf, __bf16* __restrict__ rb) {
  const int t = blockIdx.x;
  const int d = threadIdx.x;
  const int c = idx[t];
  float q = emb[(size_t)c * RVQ_EDIM + d];
  size_t i = (size_t)t * RVQ_EDIM + d;
  float r = rf[i] - q;
  rf[i] = r;
  rb[i] = (__bf16)r;
}

// ---------------------------------------------------------------------------
// finalize: z_q = z - r_final; z_q_st == z_q numerically;
// loss = (1 + BETA) * mean((z_q - z)^2) = 1.25 * mean(r^2).
// ---------------------------------------------------------------------------
__global__ __launch_bounds__(256)
void rvq_finalize(const float* __restrict__ z, const float* __restrict__ rf,
                  float* __restrict__ out) {
  size_t i = (size_t)blockIdx.x * 256 + threadIdx.x;
  float r = rf[i];
  out[1 + i] = z[i] - r;
  float s = r * r;
#pragma unroll
  for (int off = 16; off > 0; off >>= 1) s += __shfl_down(s, off, 32);
  __shared__ float ps[8];
  if ((threadIdx.x & 31) == 0) ps[threadIdx.x >> 5] = s;
  __syncthreads();
  if (threadIdx.x == 0) {
    float t = 0.0f;
#pragma unroll
    for (int w = 0; w < 8; ++w) t += ps[w];
    atomicAdd(out, t * (1.25f / 8388608.0f));  // 1.25 / (32768*256)
  }
}

// ---------------------------------------------------------------------------
extern "C" void kernel_launch(void* const* d_in, const int* in_sizes, int n_in,
                              void* d_out, int out_size, void* d_ws, size_t ws_size,
                              hipStream_t stream) {
  (void)in_sizes; (void)n_in; (void)out_size; (void)ws_size;
  const float* z   = (const float*)d_in[0];
  const float* emb = (const float*)d_in[1];
  float* out = (float*)d_out;

  char* ws = (char*)d_ws;
  float*  rf  = (float*)ws;                                   // 32 MB fp32 residual
  __bf16* rb  = (__bf16*)(ws + 33554432);                     // 16 MB bf16 residual
  __bf16* ebb = (__bf16*)(ws + 33554432 + 16777216);          //  4 MB bf16 codebook
  float*  esq = (float*)(ws + 33554432 + 16777216 + 4194304); // 32 KB ||e||^2
  int*    idx = (int*)(ws + 33554432 + 16777216 + 4194304 + 32768); // 128 KB

  rvq_prep_emb<<<RVQ_NE, 256, 0, stream>>>(emb, ebb, esq);
  rvq_prep_r<<<(RVQ_NTOK * RVQ_EDIM) / 256, 256, 0, stream>>>(z, rf, rb, out);

  for (int d = 0; d < RVQ_DEPTH; ++d) {
    rvq_argmin<<<RVQ_NTOK / 128, 128, 0, stream>>>(rb, ebb, esq, idx);
    rvq_update<<<RVQ_NTOK, 256, 0, stream>>>(emb, idx, rf, rb);
  }

  rvq_finalize<<<(RVQ_NTOK * RVQ_EDIM) / 256, 256, 0, stream>>>(z, rf, out);
}
